// x3_18150531793214
// MI455X (gfx1250) — compile-verified, matrix operand order checked by reading, not run on hardware
//
#include <hip/hip_runtime.h>
#include <hip/hip_bf16.h>
#include <math.h>

// ---------------------------------------------------------------------------
// Types / helpers
// ---------------------------------------------------------------------------
typedef unsigned short bfraw;                                   // raw bf16 bits
typedef __attribute__((ext_vector_type(16))) __bf16 v16bf;      // WMMA A/B frag
typedef __attribute__((ext_vector_type(8)))  float  v8f;        // WMMA C/D frag

union FragBF { v16bf v; float4 q[2]; };

__device__ __forceinline__ float b2f(bfraw v) {
    return __uint_as_float(((unsigned int)v) << 16);
}
__device__ __forceinline__ bfraw f2b(float f) {
    unsigned int u = __float_as_uint(f);
    unsigned int r = (u + 0x7FFFu + ((u >> 16) & 1u)) >> 16;    // RNE
    return (bfraw)r;
}
__device__ __forceinline__ float sigmoidf(float x) {
    return 1.f / (1.f + __expf(-x));
}

// Load one WMMA fragment given a uniform base pointer and a 32-bit element
// offset (lane part + compile-time tile part) -> saddr + voffset codegen.
__device__ __forceinline__ v16bf load_frag_off(const bfraw* __restrict__ base,
                                               unsigned off) {
    FragBF f;
    f.q[0] = *(const float4*)(base + off);        // K = k0+8h .. +7
    f.q[1] = *(const float4*)(base + off + 16);   // K = k0+16+8h .. +7
    return f.v;
}

#define WMMA_BF16(a, b, c) \
    __builtin_amdgcn_wmma_f32_16x16x32_bf16(false, (a), false, (b), (short)0, (c), false, false)

// ---------------------------------------------------------------------------
// Constants
// ---------------------------------------------------------------------------
#define S_LEN 256
#define BATCH 128
#define DRAW  1712          // true concat dim
#define DPAD  1728          // padded to multiple of 32 for K-loop
#define HID   256
#define GATES 1024          // 4*HID

// ---------------------------------------------------------------------------
// 0) f32 -> bf16 convert with optional K padding (zero pad cols)
// ---------------------------------------------------------------------------
__global__ void cvt_pad_kernel(const float* __restrict__ src, bfraw* __restrict__ dst,
                               int rows, int K, int ldK) {
    int idx = blockIdx.x * 256 + threadIdx.x;
    int total = rows * ldK;
    if (idx >= total) return;
    int r = idx / ldK, k = idx - r * ldK;
    float v = (k < K) ? src[(size_t)r * K + k] : 0.f;
    dst[idx] = f2b(v);
}

// ---------------------------------------------------------------------------
// 1) embedding gather + positional add -> v bf16 [S*B][DPAD]
// ---------------------------------------------------------------------------
struct EmbPtrs { const float* emb[25]; const float* pos[25]; };

__global__ void embed_kernel(const int* __restrict__ x, EmbPtrs p,
                             bfraw* __restrict__ v) {
    int sb = blockIdx.x;                 // 0 .. S*B-1
    int s = sb >> 7;                     // /128
    const int* xr = x + (size_t)sb * 25;
    bfraw* out = v + (size_t)sb * DPAD;
    for (int d = threadIdx.x; d < DPAD; d += 256) {
        float val = 0.f;
        if (d < DRAW) {
            int l, j, dim;
            if (d < 1200) { l = d / 50; j = d - 50 * l; dim = 50; }
            else          { l = 24;     j = d - 1200;   dim = 512; }
            int id = xr[l];
            val = p.emb[l][(size_t)id * dim + j] + p.pos[l][(size_t)s * dim + j];
        }
        out[d] = f2b(val);
    }
}

// ---------------------------------------------------------------------------
// 2) vs[b][d] = sum_s v[s][b][d]  (f32 accumulate, bf16 out)
// ---------------------------------------------------------------------------
__global__ void vs_kernel(const bfraw* __restrict__ v, bfraw* __restrict__ vs) {
    int idx = blockIdx.x * 256 + threadIdx.x;
    if (idx >= BATCH * DPAD) return;
    int b = idx / DPAD, d = idx - b * DPAD;
    float acc = 0.f;
    for (int s = 0; s < S_LEN; ++s)
        acc += b2f(v[((size_t)s * BATCH + b) * DPAD + d]);
    vs[idx] = f2b(acc);
}

// ---------------------------------------------------------------------------
// 3) WMMA GEMM: out = relu(A[M x LDK] * W[N x LDK]^T + bias)
//    LDK compile-time so tile offsets fold into the 24-bit inst offset.
//    block = 128(M) x 128(N) macrotile, 8 waves, wave w -> 16-row strip.
//    Per k-chunk: A + 4 B frags per group (clause + partial waits), 2 groups.
//    __launch_bounds__(256,1) raises the VGPR budget -> no accumulator spills.
// ---------------------------------------------------------------------------
template <int LDK>
__global__ __launch_bounds__(256, 1)
void gemm_relu_kernel(const bfraw* __restrict__ A,
                      const bfraw* __restrict__ W,
                      const float* __restrict__ bias,
                      bfraw* __restrict__ outB,   // bf16 out (or null)
                      float* __restrict__ outF,   // f32  out (or null)
                      int ldO) {
    int wv = threadIdx.x >> 5, lane = threadIdx.x & 31;
    int r = lane & 15, half = lane >> 4;
    int m0 = blockIdx.y * 128 + wv * 16;
    int n0 = blockIdx.x * 128;

    const bfraw* Abase = A + (size_t)m0 * LDK;
    const bfraw* Wbase = W + (size_t)n0 * LDK;
    unsigned laneOff = (unsigned)(r * LDK + 8 * half);   // same layout for A and W

    v8f acc[8] = {};
    for (int k = 0; k < LDK; k += 32) {
        unsigned o = laneOff + (unsigned)k;
        v16bf a = load_frag_off(Abase, o);
        v16bf b0[4], b1[4];
        #pragma unroll
        for (int t = 0; t < 4; ++t)
            b0[t] = load_frag_off(Wbase, o + (unsigned)(16 * t * LDK));
        #pragma unroll
        for (int t = 0; t < 4; ++t)
            b1[t] = load_frag_off(Wbase, o + (unsigned)(16 * (t + 4) * LDK));
        #pragma unroll
        for (int t = 0; t < 4; ++t)
            acc[t] = WMMA_BF16(a, b0[t], acc[t]);
        #pragma unroll
        for (int t = 0; t < 4; ++t)
            acc[t + 4] = WMMA_BF16(a, b1[t], acc[t + 4]);
    }

    #pragma unroll
    for (int t = 0; t < 8; ++t) {
        int n = n0 + 16 * t + r;
        float bv = bias ? bias[n] : 0.f;
        #pragma unroll
        for (int i = 0; i < 8; ++i) {
            int m = m0 + i + 8 * half;
            float g = acc[t][i] + bv;
            g = g > 0.f ? g : 0.f;
            if (outB) outB[(size_t)m * ldO + n] = f2b(g);
            else      outF[(size_t)m * ldO + n] = g;
        }
    }
}

// ---------------------------------------------------------------------------
// 4) persistent LSTM scan: 8 workgroups x 16 batch rows, 4 layers, 256 steps
//    LDS: gates [16][1024] f32 | h [4][16][256] bf16 | x [16][128] bf16
//         | c [4][16][256] f32   (cell state in LDS to free VGPRs)
// ---------------------------------------------------------------------------
struct LstmPtrs {
    const bfraw* wih[4];   // [1024][in]  bf16   (in = 128 layer0 else 256)
    const bfraw* whh[4];   // [1024][256] bf16
    const float* bih[4];
    const float* bhh[4];
};

__global__ __launch_bounds__(256, 1)
void lstm_kernel(const bfraw* __restrict__ s,    // [S*B][128] bf16
                 const bfraw* __restrict__ h0,   // [B][1024]  bf16
                 const float* __restrict__ c0,   // [B][1024]  f32
                 LstmPtrs w,
                 float* __restrict__ y) {        // [B][1024]  f32
    extern __shared__ char smem[];
    float* gates = (float*)smem;                                    // 64 KB
    bfraw* hls   = (bfraw*)(smem + 65536);                          // 16 KB
    bfraw* xls   = (bfraw*)(smem + 65536 + 16384);                  //  4 KB
    float* cls   = (float*)(smem + 65536 + 16384 + 4096);           // 64 KB

    int tid = threadIdx.x, wv = tid >> 5, lane = tid & 31;
    int r = lane & 15, half = lane >> 4;
    int b0 = blockIdx.x * 16;
    int n0 = wv * 128;

    // init h (bf16 LDS) and c (f32 LDS); thread tid owns hidden column j == tid
    for (int i = tid; i < 4 * 16 * HID; i += 256) {
        int l = i >> 12, rem = i & 4095, m = rem >> 8, j = rem & 255;
        hls[i] = h0[(size_t)(b0 + m) * GATES + (l << 8) + j];
        cls[i] = c0[(size_t)(b0 + m) * GATES + (l << 8) + j];
    }
    // hoist biases out of the scan (PyTorch gate order i,f,g,o)
    float bI[4], bF[4], bG[4], bO[4];
    #pragma unroll
    for (int l = 0; l < 4; ++l) {
        bI[l] = w.bih[l][tid]       + w.bhh[l][tid];
        bF[l] = w.bih[l][256 + tid] + w.bhh[l][256 + tid];
        bG[l] = w.bih[l][512 + tid] + w.bhh[l][512 + tid];
        bO[l] = w.bih[l][768 + tid] + w.bhh[l][768 + tid];
    }
    __syncthreads();

    // per-thread async-copy addressing for the layer-0 input tile:
    // 16 rows x 256B = 256 chunks of 16B, exactly one per thread
    int xm = tid >> 4, xc = tid & 15;
    unsigned xldsOff = (unsigned)(size_t)(xls + xm * 128 + xc * 8);
    const bfraw* xsrcBase = s + ((size_t)(b0 + xm)) * 128 + xc * 8;

    for (int t = 0; t < S_LEN; ++t) {
        // async global -> LDS copy of x tile (ASYNCcnt-tracked)
        {
            const bfraw* g = xsrcBase + (size_t)t * BATCH * 128;
            asm volatile("global_load_async_to_lds_b128 %0, %1, off"
                         :: "v"(xldsOff), "v"(g) : "memory");
            asm volatile("s_wait_asynccnt 0x0" ::: "memory");
        }
        __syncthreads();

        #pragma unroll
        for (int l = 0; l < 4; ++l) {
            const int K1 = (l == 0) ? 128 : 256;
            const bfraw* A1 = (l == 0) ? xls : (hls + (l - 1) * 16 * HID);
            const bfraw* A2 = hls + l * 16 * HID;
            const bfraw* Wih_n = w.wih[l] + (size_t)n0 * K1;
            const bfraw* Whh_n = w.whh[l] + (size_t)n0 * HID;
            unsigned lane1 = (unsigned)(r * K1 + 8 * half);
            unsigned lane2 = (unsigned)(r * HID + 8 * half);

            v8f acc[8] = {};
            for (int k = 0; k < K1; k += 32) {
                unsigned o = lane1 + (unsigned)k;
                v16bf a = load_frag_off(A1, o);
                v16bf b0f[4], b1f[4];
                #pragma unroll
                for (int tt = 0; tt < 4; ++tt)
                    b0f[tt] = load_frag_off(Wih_n, o + (unsigned)(16 * tt * K1));
                #pragma unroll
                for (int tt = 0; tt < 4; ++tt)
                    b1f[tt] = load_frag_off(Wih_n, o + (unsigned)(16 * (tt + 4) * K1));
                #pragma unroll
                for (int tt = 0; tt < 4; ++tt)
                    acc[tt] = WMMA_BF16(a, b0f[tt], acc[tt]);
                #pragma unroll
                for (int tt = 0; tt < 4; ++tt)
                    acc[tt + 4] = WMMA_BF16(a, b1f[tt], acc[tt + 4]);
            }
            for (int k = 0; k < HID; k += 32) {
                unsigned o = lane2 + (unsigned)k;
                v16bf a = load_frag_off(A2, o);
                v16bf b0f[4], b1f[4];
                #pragma unroll
                for (int tt = 0; tt < 4; ++tt)
                    b0f[tt] = load_frag_off(Whh_n, o + (unsigned)(16 * tt * HID));
                #pragma unroll
                for (int tt = 0; tt < 4; ++tt)
                    b1f[tt] = load_frag_off(Whh_n, o + (unsigned)(16 * (tt + 4) * HID));
                #pragma unroll
                for (int tt = 0; tt < 4; ++tt)
                    acc[tt] = WMMA_BF16(a, b0f[tt], acc[tt]);
                #pragma unroll
                for (int tt = 0; tt < 4; ++tt)
                    acc[tt + 4] = WMMA_BF16(a, b1f[tt], acc[tt + 4]);
            }
            #pragma unroll
            for (int tt = 0; tt < 8; ++tt)
                #pragma unroll
                for (int i = 0; i < 8; ++i)
                    gates[(i + 8 * half) * GATES + n0 + 16 * tt + r] = acc[tt][i];
            __syncthreads();

            // elementwise cell update; thread owns hidden column j = tid
            #pragma unroll
            for (int m = 0; m < 16; ++m) {
                float gi = gates[m * GATES + tid]       + bI[l];
                float gf = gates[m * GATES + 256 + tid] + bF[l];
                float gg = gates[m * GATES + 512 + tid] + bG[l];
                float go = gates[m * GATES + 768 + tid] + bO[l];
                int ci = (l * 16 + m) * HID + tid;
                float c = sigmoidf(gf) * cls[ci] + sigmoidf(gi) * tanhf(gg);
                cls[ci] = c;
                float h = sigmoidf(go) * tanhf(c);
                hls[l * 16 * HID + m * HID + tid] = f2b(h);
                if (t == S_LEN - 1)
                    y[(size_t)(b0 + m) * GATES + (l << 8) + tid] = h;
            }
            __syncthreads();
        }
    }
}

// ---------------------------------------------------------------------------
// host launcher
// ---------------------------------------------------------------------------
extern "C" void kernel_launch(void* const* d_in, const int* in_sizes, int n_in,
                              void* d_out, int out_size, void* d_ws, size_t ws_size,
                              hipStream_t stream) {
    (void)in_sizes; (void)n_in; (void)out_size; (void)ws_size;

    const int* x = (const int*)d_in[0];
    EmbPtrs ep;
    for (int i = 0; i < 25; ++i) ep.emb[i] = (const float*)d_in[1 + i];
    for (int i = 0; i < 25; ++i) ep.pos[i] = (const float*)d_in[26 + i];
    const float* f01_w = (const float*)d_in[51];
    const float* f01_b = (const float*)d_in[52];
    const float* f02_w = (const float*)d_in[53];
    const float* f02_b = (const float*)d_in[54];
    const float* f03_w = (const float*)d_in[55];
    const float* f03_b = (const float*)d_in[56];

    // workspace carve (256B aligned)
    size_t off = 0;
    auto carve = [&](size_t bytes) {
        void* p = (char*)d_ws + off;
        off += (bytes + 255) & ~(size_t)255;
        return p;
    };
    bfraw* v_bf   = (bfraw*)carve((size_t)S_LEN * BATCH * DPAD * 2);
    bfraw* s_bf   = (bfraw*)carve((size_t)S_LEN * BATCH * 128 * 2);
    bfraw* vs_bf  = (bfraw*)carve((size_t)BATCH * DPAD * 2);
    bfraw* w01    = (bfraw*)carve((size_t)128 * DPAD * 2);
    bfraw* w02    = (bfraw*)carve((size_t)GATES * DPAD * 2);
    bfraw* w03    = (bfraw*)carve((size_t)GATES * DPAD * 2);
    bfraw* h0_bf  = (bfraw*)carve((size_t)BATCH * GATES * 2);
    float* c0_f   = (float*)carve((size_t)BATCH * GATES * 4);
    LstmPtrs lp;
    for (int l = 0; l < 4; ++l) {
        int in = (l == 0) ? 128 : 256;
        lp.wih[l] = (bfraw*)carve((size_t)GATES * in * 2);
        lp.whh[l] = (bfraw*)carve((size_t)GATES * HID * 2);
        lp.bih[l] = (const float*)d_in[65 + l];
        lp.bhh[l] = (const float*)d_in[69 + l];
    }

    auto cvt = [&](const float* src, bfraw* dst, int rows, int K, int ldK) {
        int total = rows * ldK;
        cvt_pad_kernel<<<(total + 255) / 256, 256, 0, stream>>>(src, dst, rows, K, ldK);
    };

    // weight conversion f32 -> bf16 (K padded to 1728 where needed)
    cvt(f01_w, w01, 128, DRAW, DPAD);
    cvt(f02_w, w02, GATES, DRAW, DPAD);
    cvt(f03_w, w03, GATES, DRAW, DPAD);
    for (int l = 0; l < 4; ++l) {
        int in = (l == 0) ? 128 : 256;
        cvt((const float*)d_in[57 + l], (bfraw*)lp.wih[l], GATES, in, in);
        cvt((const float*)d_in[61 + l], (bfraw*)lp.whh[l], GATES, HID, HID);
    }

    // embeddings + positional add
    embed_kernel<<<S_LEN * BATCH, 256, 0, stream>>>(x, ep, v_bf);

    // vs = sum over time
    vs_kernel<<<(BATCH * DPAD + 255) / 256, 256, 0, stream>>>(v_bf, vs_bf);

    // s = relu(v @ f01^T + b)   M=32768 N=128 K=1728
    gemm_relu_kernel<DPAD><<<dim3(1, S_LEN * BATCH / 128), 256, 0, stream>>>(
        v_bf, w01, f01_b, s_bf, nullptr, 128);

    // h0 = relu(vs @ f02^T + b), c0 = relu(vs @ f03^T + b)   M=128 N=1024 K=1728
    gemm_relu_kernel<DPAD><<<dim3(GATES / 128, 1), 256, 0, stream>>>(
        vs_bf, w02, f02_b, h0_bf, nullptr, GATES);
    gemm_relu_kernel<DPAD><<<dim3(GATES / 128, 1), 256, 0, stream>>>(
        vs_bf, w03, f03_b, nullptr, c0_f, GATES);

    // persistent LSTM scan, batch-split across 8 workgroups
    size_t lds_bytes = 65536 + 16384 + 4096 + 65536;   // gates | h | x | c
    lstm_kernel<<<BATCH / 16, 256, lds_bytes, stream>>>(
        s_bf, h0_bf, c0_f, lp, (float*)d_out);
}
